// CachedHead_28183575396812
// MI455X (gfx1250) — matrix operand order repaired and checked by reading, not compiled, not run on hardware
//
#include <hip/hip_runtime.h>
#include <math.h>

typedef __attribute__((ext_vector_type(16))) _Float16 v16h;
typedef __attribute__((ext_vector_type(8)))  _Float16 v8h;
typedef __attribute__((ext_vector_type(4)))  _Float16 v4h;
typedef __attribute__((ext_vector_type(8)))  float    v8f;

#define T_SEQ 4096
#define CDIM  1024
#define DDIM  64
#define BATCH 4

static __device__ __forceinline__ v8f wmma16(v16h a, v16h b, v8f c) {
  // D = A(16x32 f16) * B(32x16 f16) + C(16x16 f32)
  return __builtin_amdgcn_wmma_f32_16x16x32_f16(false, a, false, b, (short)0, c,
                                                false, false);
}

// Assemble a v16h A-fragment from two contiguous 8-half (16B) chunks.
static __device__ __forceinline__ v16h ld2x8(const _Float16* p0, const _Float16* p1) {
  v8h a = *(const v8h*)p0;
  v8h b = *(const v8h*)p1;
  v16h r;
#pragma unroll
  for (int i = 0; i < 8; ++i) { r[i] = a[i]; r[8 + i] = b[i]; }
  return r;
}

// ---------------- kernel 1: x fp32 -> f16 ----------------
__global__ void cvt_x_f32_to_f16(const float* __restrict__ x,
                                 _Float16* __restrict__ xh, int n4) {
  int i = blockIdx.x * blockDim.x + threadIdx.x;
  if (i < n4) {
    float4 f = ((const float4*)x)[i];
    v4h h;
    h[0] = (_Float16)f.x; h[1] = (_Float16)f.y;
    h[2] = (_Float16)f.z; h[3] = (_Float16)f.w;
    *(v4h*)(xh + (size_t)i * 4) = h;
  }
}

// ---------------- kernel 2: pack W into B-fragment layout ----------------
// wpk[w][kb][nt][lane][16]: lane<16 -> column n=nt*16+lane, K = kb*32 + 0..15
//                           lane>=16 -> same column set,    K = kb*32 + 16..31
__global__ void pack_w_kernel(const float* __restrict__ Wq,
                              const float* __restrict__ Wk,
                              const float* __restrict__ Wv,
                              _Float16* __restrict__ wpk) {
  int tid = blockIdx.x * 256 + threadIdx.x;       // 12288 total
  int l  = tid & 31;
  int nt = (tid >> 5) & 3;
  int kb = (tid >> 7) & 31;
  int w  = tid >> 12;                             // 0=q 1=k 2=v
  const float* W = (w == 0) ? Wq : (w == 1) ? Wk : Wv;
  int n     = nt * 16 + (l & 15);
  int kbase = kb * 32 + 16 * (l >> 4);
  _Float16* dst = wpk + ((size_t)((w * 32 + kb) * 4 + nt)) * 512 + (size_t)l * 16;
#pragma unroll
  for (int j = 0; j < 16; ++j)
    dst[j] = (_Float16)W[(size_t)(kbase + j) * DDIM + n];
}

// ---------------- kernel 3: projections q,k,v = x @ W ----------------
// grid (1024, 3); one wave computes a 16x64 tile. y: 0->q, 1->k (row-major),
// 2->v (stored transposed [D][T] for the PV B-fragments).
__global__ void __launch_bounds__(32)
proj_gemm(const _Float16* __restrict__ xh, const _Float16* __restrict__ wpk,
          _Float16* __restrict__ qh, _Float16* __restrict__ kh,
          _Float16* __restrict__ vTh) {
  const int rt   = blockIdx.x;          // row tile (16 rows of B*T)
  const int w    = blockIdx.y;
  const int lane = threadIdx.x;
  const int lm = lane & 15, hi = lane >> 4;

  v8f acc[4] = {};
  const _Float16* arow = xh + (size_t)(rt * 16 + lm) * CDIM;
  const _Float16* bw   = wpk + (size_t)w * 32 * 4 * 512;

  for (int kb = 0; kb < 32; ++kb) {
    const _Float16* ap = arow + kb * 32 + hi * 16;    // K 0..15 + 16*hi
    v16h a = *(const v16h*)ap;
#pragma unroll
    for (int nt = 0; nt < 4; ++nt) {
      v16h b = *(const v16h*)(bw + ((size_t)(kb * 4 + nt)) * 512 + (size_t)lane * 16);
      acc[nt] = wmma16(a, b, acc[nt]);
    }
  }

  if (w < 2) {
    _Float16* dst = (w == 0) ? qh : kh;
#pragma unroll
    for (int nt = 0; nt < 4; ++nt)
#pragma unroll
      for (int r = 0; r < 8; ++r)
        dst[(size_t)(rt * 16 + r + 8 * hi) * DDIM + nt * 16 + lm] =
            (_Float16)acc[nt][r];
  } else {
    const int b  = rt >> 8;                    // 256 row-tiles per batch
    const int t0 = (rt & 255) * 16 + 8 * hi;   // 8 consecutive t per lane
#pragma unroll
    for (int nt = 0; nt < 4; ++nt) {
      v8h h;
#pragma unroll
      for (int r = 0; r < 8; ++r) h[r] = (_Float16)acc[nt][r];
      *(v8h*)(vTh + (size_t)b * DDIM * T_SEQ +
              (size_t)(nt * 16 + lm) * T_SEQ + t0) = h;
    }
  }
}

// ---------------- kernel 4: zero strict upper triangle of wei ----------------
__global__ void zero_upper(float* __restrict__ wei) {
  const int b   = blockIdx.x >> 12;
  const int row = blockIdx.x & 4095;
  float* wr = wei + (size_t)b * T_SEQ * T_SEQ + (size_t)row * T_SEQ;
  for (int j = row + 1 + threadIdx.x; j < T_SEQ; j += blockDim.x) wr[j] = 0.0f;
}

// ---------------- kernel 5: causal attention, one wave per 16-row q tile ----
__global__ void __launch_bounds__(32)
attn_kernel(const _Float16* __restrict__ qh, const _Float16* __restrict__ kh,
            const _Float16* __restrict__ vTh, float* __restrict__ out,
            float* __restrict__ wei) {
  __shared__ _Float16 ldsA[16 * 32];   // P staged as f16 for A-fragments
  __shared__ float    ldsP[16 * 32];   // P fp32 for coalesced wei writes

  const int qt   = blockIdx.x;         // 0..255
  const int b    = blockIdx.y;
  const int lane = threadIdx.x;
  const int lm = lane & 15, hi = lane >> 4;

  const _Float16* qb = qh  + (size_t)b * T_SEQ * DDIM;
  const _Float16* kb = kh  + (size_t)b * T_SEQ * DDIM;
  const _Float16* vT = vTh + (size_t)b * DDIM * T_SEQ;
  float* weib = wei + (size_t)b * T_SEQ * T_SEQ;

  const _Float16* qr = qb + (size_t)(qt * 16 + lm) * DDIM;
  const v16h aq0 = *(const v16h*)(qr + hi * 16);
  const v16h aq1 = *(const v16h*)(qr + 32 + hi * 16);

  float rm[8], sm[8], inv[8];
#pragma unroll
  for (int r = 0; r < 8; ++r) { rm[r] = -1e30f; sm[r] = 0.0f; }

  // ---- pass 1: row max (elementwise, reduce once at the end) ----
  for (int nt = 0; nt <= qt; ++nt) {
    const _Float16* kr = kb + (size_t)(nt * 16 + lm) * DDIM;
    v16h b0 = *(const v16h*)(kr + hi * 16);
    v16h b1 = *(const v16h*)(kr + 32 + hi * 16);
    v8f s = {};
    s = wmma16(aq0, b0, s);
    s = wmma16(aq1, b1, s);
#pragma unroll
    for (int r = 0; r < 8; ++r) {
      const int row = qt * 16 + r + 8 * hi, col = nt * 16 + lm;
      float v = s[r] * 0.125f;                    // 1/sqrt(64)
      v = (col > row) ? -1e30f : v;
      rm[r] = fmaxf(rm[r], v);
    }
  }
#pragma unroll
  for (int r = 0; r < 8; ++r)
    for (int mk = 1; mk < 16; mk <<= 1)
      rm[r] = fmaxf(rm[r], __shfl_xor(rm[r], mk));

  // ---- pass 2: row sum of exp ----
  for (int nt = 0; nt <= qt; ++nt) {
    const _Float16* kr = kb + (size_t)(nt * 16 + lm) * DDIM;
    v16h b0 = *(const v16h*)(kr + hi * 16);
    v16h b1 = *(const v16h*)(kr + 32 + hi * 16);
    v8f s = {};
    s = wmma16(aq0, b0, s);
    s = wmma16(aq1, b1, s);
#pragma unroll
    for (int r = 0; r < 8; ++r) {
      const int row = qt * 16 + r + 8 * hi, col = nt * 16 + lm;
      float v = s[r] * 0.125f;
      v = (col > row) ? -1e30f : v;
      sm[r] += __expf(v - rm[r]);
    }
  }
#pragma unroll
  for (int r = 0; r < 8; ++r) {
    for (int mk = 1; mk < 16; mk <<= 1) sm[r] += __shfl_xor(sm[r], mk);
    inv[r] = 1.0f / sm[r];
  }

  // ---- pass 3: emit normalized wei and accumulate out = P @ V ----
  v8f acc[4] = {};
  for (int kp = 0; kp * 2 <= qt; ++kp) {
    const int nt0 = 2 * kp;
    const int present1 = (nt0 + 1 <= qt);
    __syncthreads();
#pragma unroll
    for (int t = 0; t < 2; ++t) {
      const int nt = nt0 + t;
      if (t == 1 && !present1) {
        // zero-pad absent half of the A fragment
#pragma unroll
        for (int r = 0; r < 8; ++r)
          ldsA[(r + 8 * hi) * 32 + 16 + lm] = (_Float16)0.0f;
        break;
      }
      const _Float16* kr = kb + (size_t)(nt * 16 + lm) * DDIM;
      v16h b0 = *(const v16h*)(kr + hi * 16);
      v16h b1 = *(const v16h*)(kr + 32 + hi * 16);
      v8f s = {};
      s = wmma16(aq0, b0, s);
      s = wmma16(aq1, b1, s);
#pragma unroll
      for (int r = 0; r < 8; ++r) {
        const int row = qt * 16 + r + 8 * hi, col = nt * 16 + lm;
        float v = s[r] * 0.125f;
        v = (col > row) ? -1e30f : v;
        float p = __expf(v - rm[r]) * inv[r];
        const int m = r + 8 * hi;
        ldsA[m * 32 + lm + 16 * t] = (_Float16)p;
        ldsP[m * 32 + lm + 16 * t] = p;
      }
    }
    __syncthreads();

    // coalesced wei writes: 16 rows x (32 or 16) floats
    {
      float* wrow = weib + (size_t)(qt * 16) * T_SEQ + nt0 * 16;
      const int ncol = present1 ? 32 : 16;
      for (int row = 0; row < 16; ++row)
        if (lane < ncol)
          wrow[(size_t)row * T_SEQ + lane] = ldsP[row * 32 + lane];
    }

    // PV: A = P (16x32), B = V^T rows (contiguous in k-index)
    v16h pa = *(const v16h*)&ldsA[lm * 32 + 16 * hi];
#pragma unroll
    for (int dt = 0; dt < 4; ++dt) {
      const _Float16* vr = vT + (size_t)(dt * 16 + lm) * T_SEQ + kp * 32 + hi * 16;
      v16h vb = *(const v16h*)vr;
      acc[dt] = wmma16(pa, vb, acc[dt]);
    }
  }

#pragma unroll
  for (int dt = 0; dt < 4; ++dt)
#pragma unroll
    for (int r = 0; r < 8; ++r)
      out[(size_t)b * T_SEQ * DDIM + (size_t)(qt * 16 + r + 8 * hi) * DDIM +
          dt * 16 + lm] = acc[dt][r];
}

extern "C" void kernel_launch(void* const* d_in, const int* in_sizes, int n_in,
                              void* d_out, int out_size, void* d_ws, size_t ws_size,
                              hipStream_t stream) {
  (void)in_sizes; (void)n_in; (void)out_size; (void)ws_size;
  const float* x  = (const float*)d_in[0];
  const float* Wk = (const float*)d_in[1];
  const float* Wq = (const float*)d_in[2];
  const float* Wv = (const float*)d_in[3];
  float* out = (float*)d_out;
  float* wei = out + (size_t)BATCH * T_SEQ * DDIM;

  // workspace layout (f16 elements): ~40 MB total
  _Float16* xh  = (_Float16*)d_ws;
  _Float16* wpk = xh  + (size_t)BATCH * T_SEQ * CDIM;   // 16,777,216
  _Float16* qh  = wpk + (size_t)3 * 32 * 4 * 512;       //    196,608
  _Float16* kh  = qh  + (size_t)BATCH * T_SEQ * DDIM;   //  1,048,576
  _Float16* vTh = kh  + (size_t)BATCH * T_SEQ * DDIM;   //  1,048,576

  const int n4 = BATCH * T_SEQ * CDIM / 4;
  cvt_x_f32_to_f16<<<(n4 + 255) / 256, 256, 0, stream>>>(x, xh, n4);
  pack_w_kernel<<<48, 256, 0, stream>>>(Wq, Wk, Wv, wpk);
  proj_gemm<<<dim3(1024, 3), 32, 0, stream>>>(xh, wpk, qh, kh, vTh);
  zero_upper<<<BATCH * T_SEQ, 256, 0, stream>>>(wei);
  attn_kernel<<<dim3(256, BATCH), 32, 0, stream>>>(qh, kh, vTh, out, wei);
}
